// MemoryEnhancedModule_46557445488996
// MI455X (gfx1250) — compile-verified, matrix operand order checked by reading, not run on hardware
//
#include <hip/hip_runtime.h>

typedef __attribute__((ext_vector_type(16))) _Float16 v16h;
typedef __attribute__((ext_vector_type(8)))  float    v8f;
typedef __attribute__((ext_vector_type(4)))  _Float16 v4h;

#define NTOK  16384   // 8 * 2048 tokens
#define DIM   1024
#define MSLOT 64

#define AS3 __attribute__((address_space(3)))

// Feature-detect the CDNA5 async global->LDS path (builtin exists on this
// toolchain; we drive the instruction via inline asm to avoid its
// unspellable cuda_device-AS pointer prototype).
#if __has_builtin(__builtin_amdgcn_global_load_async_to_lds_b128)
#define MEMX_ASYNC 1
#endif

union Frag16 { v16h v; float4 q[2]; };

// Load one 16x32 f16 WMMA operand fragment for this lane.
// rowbase = pointer to this lane's row (K contiguous), koff = (lane&16)?8:0.
static __device__ inline v16h load_frag(const _Float16* rowbase, int koff) {
  Frag16 f;
  f.q[0] = *(const float4*)(rowbase + koff);       // K koff..koff+7
  f.q[1] = *(const float4*)(rowbase + 16 + koff);  // K 16+koff..23+koff
  return f.v;
}

static __device__ inline void memx_wait_async() {
#ifdef MEMX_ASYNC
  asm volatile("s_wait_asynccnt 0x0" ::: "memory");
#endif
}

// 16-byte global -> LDS copy (ASYNCcnt-tracked DMA on CDNA5, sync fallback)
static __device__ inline void memx_cp16(const _Float16* g, _Float16* l) {
#ifdef MEMX_ASYNC
  unsigned ldsAddr = (unsigned)(unsigned long long)(AS3 _Float16*)l;
  asm volatile("global_load_async_to_lds_b128 %0, %1, off"
               :: "v"(ldsAddr), "v"(g)
               : "memory");
#else
  *(float4*)l = *(const float4*)g;
#endif
}

// ---------------- Phase 0a: P = mem @ Wq (f32 -> f16), c[m] = bq . mem[m] ----
__global__ void memx_prep_P(const float* __restrict__ mem,
                            const float* __restrict__ Wq,
                            const float* __restrict__ bq,
                            _Float16* __restrict__ Ph,
                            float* __restrict__ cm) {
  int idx = blockIdx.x * blockDim.x + threadIdx.x;   // 0 .. 64*1024-1
  int m = idx >> 10;
  int d = idx & 1023;
  const float* mr = mem + (size_t)m * DIM;
  float acc = 0.f;
  for (int e = 0; e < DIM; ++e)
    acc = fmaf(mr[e], Wq[(size_t)e * DIM + d], acc);   // P[m,d] = sum_e mem[m,e]*Wq[e,d]
  Ph[(size_t)m * DIM + d] = (_Float16)acc;
  if (d == 0) {
    float c = 0.f;
    for (int e = 0; e < DIM; ++e) c = fmaf(bq[e], mr[e], c);
    cm[m] = c;
  }
}

// ---------------- Phase 0b: f32 -> f16 conversion (vectorized) ---------------
__global__ void memx_cvt_f16(const float* __restrict__ in,
                             _Float16* __restrict__ out, int n4) {
  int i = blockIdx.x * blockDim.x + threadIdx.x;
  if (i < n4) {
    float4 v = ((const float4*)in)[i];
    v4h o;
    o[0] = (_Float16)v.x; o[1] = (_Float16)v.y;
    o[2] = (_Float16)v.z; o[3] = (_Float16)v.w;
    ((v4h*)out)[i] = o;
  }
}

// ---------------- Phase 1: sim = x@P^T + c, top3, softmax, mo ----------------
// One wave handles 16 tokens. 8 waves / block -> 128 tokens / block.
__global__ void memx_sim_topk_mo(const _Float16* __restrict__ xh,
                                 const _Float16* __restrict__ Ph,
                                 const float* __restrict__ cm,
                                 const float* __restrict__ memf,
                                 _Float16* __restrict__ moh) {
  __shared__ __align__(16) float simS[8][16][MSLOT];
  __shared__ float wS[8][16][3];
  __shared__ int   iS[8][16][3];

  const int lane = threadIdx.x & 31;
  const int wv   = threadIdx.x >> 5;
  const int tok0 = (blockIdx.x * 8 + wv) * 16;
  const int koff = (lane & 16) ? 8 : 0;

  const _Float16* aRow  = xh + (size_t)(tok0 + (lane & 15)) * DIM;
  const _Float16* bRow0 = Ph + (size_t)(lane & 15) * DIM;

  v8f acc[4] = {};                 // slots [0,16) [16,32) [32,48) [48,64)
  for (int k = 0; k < DIM; k += 32) {
    // load A + all four B fragments into distinct registers first so the
    // loads clause and the WMMAs don't each stall on s_wait_loadcnt 0
    v16h a = load_frag(aRow + k, koff);
    v16h b[4];
#pragma unroll
    for (int j = 0; j < 4; ++j)
      b[j] = load_frag(bRow0 + (size_t)j * 16 * DIM + k, koff);
#pragma unroll
    for (int j = 0; j < 4; ++j)
      acc[j] = __builtin_amdgcn_wmma_f32_16x16x32_f16(
          false, a, false, b[j], (short)0, acc[j], false, false);
  }

  // scatter sims to LDS, adding the bq.mem[m] bias term
  const int ncol = lane & 15;
  const int mrow = (lane < 16) ? 0 : 8;
#pragma unroll
  for (int j = 0; j < 4; ++j) {
    float cv = cm[j * 16 + ncol];
#pragma unroll
    for (int r = 0; r < 8; ++r)
      simS[wv][mrow + r][j * 16 + ncol] = acc[j][r] + cv;
  }
  __syncthreads();

  // strict top-3 (earliest index wins ties, matching jax.lax.top_k) + softmax
  if (lane < 16) {
    float v0 = -INFINITY, v1 = -INFINITY, v2 = -INFINITY;
    int   i0 = 0, i1 = 0, i2 = 0;
    for (int m = 0; m < MSLOT; ++m) {
      float v = simS[wv][lane][m];
      if (v > v0)      { v2 = v1; i2 = i1; v1 = v0; i1 = i0; v0 = v; i0 = m; }
      else if (v > v1) { v2 = v1; i2 = i1; v1 = v;  i1 = m; }
      else if (v > v2) { v2 = v;  i2 = m; }
    }
    float e1 = __expf(v1 - v0), e2 = __expf(v2 - v0);
    float inv = 1.f / (1.f + e1 + e2);
    wS[wv][lane][0] = inv; wS[wv][lane][1] = e1 * inv; wS[wv][lane][2] = e2 * inv;
    iS[wv][lane][0] = i0;  iS[wv][lane][1] = i1;       iS[wv][lane][2] = i2;
  }
  __syncthreads();

  // mo[t,:] = w0*mem[i0] + w1*mem[i1] + w2*mem[i2]  (f32 math, f16 store)
  for (int tt = 0; tt < 16; ++tt) {
    float w0 = wS[wv][tt][0], w1 = wS[wv][tt][1], w2 = wS[wv][tt][2];
    const float* m0 = memf + (size_t)iS[wv][tt][0] * DIM;
    const float* m1 = memf + (size_t)iS[wv][tt][1] * DIM;
    const float* m2 = memf + (size_t)iS[wv][tt][2] * DIM;
    _Float16* dst = moh + (size_t)(tok0 + tt) * DIM;
    for (int d = lane * 4; d < DIM; d += 128) {
      float4 a4 = *(const float4*)(m0 + d);
      float4 b4 = *(const float4*)(m1 + d);
      float4 c4 = *(const float4*)(m2 + d);
      v4h o;
      o[0] = (_Float16)(w0 * a4.x + w1 * b4.x + w2 * c4.x);
      o[1] = (_Float16)(w0 * a4.y + w1 * b4.y + w2 * c4.y);
      o[2] = (_Float16)(w0 * a4.z + w1 * b4.z + w2 * c4.z);
      o[3] = (_Float16)(w0 * a4.w + w1 * b4.w + w2 * c4.w);
      *(v4h*)(dst + d) = o;
    }
  }
}

// ---------------- Phase 2: h = [x|mo] @ Wf^T + bf  ---------------------------
// 128x128 output tile / block, K staged in 32-wide chunks through LDS with a
// double-buffered async-copy pipeline (one barrier per K-step):
//   s_wait_asynccnt 0 -> this wave's copies into buffer p are in LDS
//   __syncthreads     -> every wave's copies landed; p^1 free to overwrite
//   stage(p^1)        -> async loads for K-step k+1
//   WMMA on p
#define TM 128
#define TN 128
#define TK 32
__global__ void memx_fusion_gemm(const _Float16* __restrict__ xh,
                                 const _Float16* __restrict__ moh,
                                 const _Float16* __restrict__ wfh,
                                 const float* __restrict__ bf,
                                 float* __restrict__ hout) {
  __shared__ __align__(16) _Float16 As[2][TM * TK];  // token rows, K contiguous
  __shared__ __align__(16) _Float16 Bs[2][TN * TK];  // Wf rows (out cols), K contiguous

  const int tid  = threadIdx.x;
  const int lane = tid & 31;
  const int wv   = tid >> 5;
  const int m0   = blockIdx.x * TM;      // token base
  const int n0   = blockIdx.y * TN;      // output-column base
  const int wm   = wv & 3;               // wave grid 4 (tokens) x 2 (cols)
  const int wn   = wv >> 2;
  const int tokBase = wm * 32;           // 2 row tiles of 16
  const int colBase = wn * 64;           // 4 col tiles of 16
  const int koff = (lane & 16) ? 8 : 0;

  v8f acc[2][4] = {};

  // cooperative stage of one 128x32 A tile + 128x32 B tile:
  // 256 threads x 2 chunks each (chunk = 16B = 8 halves, 4 chunks per row)
  auto stage = [&](int p, int k0) {
    const _Float16* srcA = (k0 < DIM) ? (xh + k0) : (moh + (k0 - DIM));
#pragma unroll
    for (int c = 0; c < 2; ++c) {
      int chunk = tid * 2 + c;            // 0..511
      int row = chunk >> 2;
      int cc  = (chunk & 3) * 8;
      memx_cp16(srcA + (size_t)(m0 + row) * DIM + cc, &As[p][row * TK + cc]);
      memx_cp16(wfh + (size_t)(n0 + row) * (2 * DIM) + k0 + cc,
                &Bs[p][row * TK + cc]);
    }
  };

  stage(0, 0);
  int p = 0;
  for (int k0 = 0; k0 < 2 * DIM; k0 += TK) {
    memx_wait_async();     // this wave's copies into buffer p complete
    __syncthreads();       // all waves' copies landed; p^1 free to overwrite
    if (k0 + TK < 2 * DIM) stage(p ^ 1, k0 + TK);

    v16h a[2], b[4];
#pragma unroll
    for (int i = 0; i < 2; ++i)
      a[i] = load_frag(&As[p][(tokBase + i * 16 + (lane & 15)) * TK], koff);
#pragma unroll
    for (int j = 0; j < 4; ++j)
      b[j] = load_frag(&Bs[p][(colBase + j * 16 + (lane & 15)) * TK], koff);
#pragma unroll
    for (int i = 0; i < 2; ++i)
#pragma unroll
      for (int j = 0; j < 4; ++j)
        acc[i][j] = __builtin_amdgcn_wmma_f32_16x16x32_f16(
            false, a[i], false, b[j], (short)0, acc[i][j], false, false);
    p ^= 1;
  }

  // epilogue: + bf, store f32 h
  const int ncol = lane & 15;
  const int mrowOff = (lane < 16) ? 0 : 8;
#pragma unroll
  for (int j = 0; j < 4; ++j) {
    int col = n0 + colBase + j * 16 + ncol;
    float bv = bf[col];
#pragma unroll
    for (int i = 0; i < 2; ++i) {
      int tokT = m0 + tokBase + i * 16 + mrowOff;
#pragma unroll
      for (int r = 0; r < 8; ++r)
        hout[(size_t)(tokT + r) * DIM + col] = acc[i][j][r] + bv;
    }
  }
}

// ---------------- Phase 3: LayerNorm + ReLU (one wave per token) -------------
__global__ void memx_ln_relu(const float* __restrict__ hin,
                             const float* __restrict__ gamma,
                             const float* __restrict__ beta,
                             float* __restrict__ out) {
  const int lane = threadIdx.x & 31;
  const int wv   = threadIdx.x >> 5;
  const size_t t = (size_t)blockIdx.x * 8 + wv;
  const float* row = hin + t * DIM;
  float4 v[8];
  float s = 0.f;
#pragma unroll
  for (int i = 0; i < 8; ++i) {
    v[i] = *(const float4*)(row + i * 128 + lane * 4);
    s += v[i].x + v[i].y + v[i].z + v[i].w;
  }
#pragma unroll
  for (int off = 16; off; off >>= 1) s += __shfl_xor(s, off, 32);
  float mean = s * (1.f / 1024.f);
  float vs = 0.f;
#pragma unroll
  for (int i = 0; i < 8; ++i) {
    float dx = v[i].x - mean; vs += dx * dx;
    dx = v[i].y - mean; vs += dx * dx;
    dx = v[i].z - mean; vs += dx * dx;
    dx = v[i].w - mean; vs += dx * dx;
  }
#pragma unroll
  for (int off = 16; off; off >>= 1) vs += __shfl_xor(vs, off, 32);
  float inv = rsqrtf(vs * (1.f / 1024.f) + 1e-5f);
  float* orow = out + t * DIM;
#pragma unroll
  for (int i = 0; i < 8; ++i) {
    int col = i * 128 + lane * 4;
    float4 g = *(const float4*)(gamma + col);
    float4 be = *(const float4*)(beta + col);
    float4 o;
    o.x = fmaxf((v[i].x - mean) * inv * g.x + be.x, 0.f);
    o.y = fmaxf((v[i].y - mean) * inv * g.y + be.y, 0.f);
    o.z = fmaxf((v[i].z - mean) * inv * g.z + be.z, 0.f);
    o.w = fmaxf((v[i].w - mean) * inv * g.w + be.w, 0.f);
    *(float4*)(orow + col) = o;
  }
}

// ---------------------------------------------------------------------------
extern "C" void kernel_launch(void* const* d_in, const int* in_sizes, int n_in,
                              void* d_out, int out_size, void* d_ws, size_t ws_size,
                              hipStream_t stream) {
  const float* x    = (const float*)d_in[0];
  const float* memf = (const float*)d_in[1];
  const float* Wq   = (const float*)d_in[2];
  const float* bq   = (const float*)d_in[3];
  const float* Wf   = (const float*)d_in[4];
  const float* bf   = (const float*)d_in[5];
  const float* gamma= (const float*)d_in[6];
  const float* beta = (const float*)d_in[7];
  float* out = (float*)d_out;

  char* ws = (char*)d_ws;
  _Float16* xh  = (_Float16*)ws;  ws += (size_t)NTOK * DIM * 2;        // 32 MB
  _Float16* moh = (_Float16*)ws;  ws += (size_t)NTOK * DIM * 2;        // 32 MB
  _Float16* wfh = (_Float16*)ws;  ws += (size_t)DIM * 2 * DIM * 2;     //  4 MB
  _Float16* Ph  = (_Float16*)ws;  ws += (size_t)MSLOT * DIM * 2;       // 128 KB
  float*    cm  = (float*)ws;     ws += 256;                           // 64 slots
  float*    hws = (float*)ws;                                          // 64 MB

  // Phase 0: precompute P = mem@Wq (+ bias terms), convert x & Wf to f16
  memx_prep_P<<<(MSLOT * DIM) / 256, 256, 0, stream>>>(memf, Wq, bq, Ph, cm);
  memx_cvt_f16<<<((size_t)NTOK * DIM / 4 + 255) / 256, 256, 0, stream>>>(
      x, xh, NTOK * DIM / 4);
  memx_cvt_f16<<<((size_t)DIM * 2 * DIM / 4 + 255) / 256, 256, 0, stream>>>(
      Wf, wfh, DIM * 2 * DIM / 4);

  // Phase 1: sim -> top3 -> softmax -> memory_output (f16)
  memx_sim_topk_mo<<<NTOK / 128, 256, 0, stream>>>(xh, Ph, cm, memf, moh);

  // Phase 2: h = [x|mo] @ Wf^T + bf
  dim3 ggrid(NTOK / TM, DIM / TN);
  memx_fusion_gemm<<<ggrid, 256, 0, stream>>>(xh, moh, wfh, bf, hws);

  // Phase 3: LayerNorm + ReLU
  memx_ln_relu<<<NTOK / 8, 256, 0, stream>>>(hws, gamma, beta, out);
}
